// Model_30391188587002
// MI455X (gfx1250) — matrix operand order repaired and checked by reading, compiled, not run on hardware
//
#include <hip/hip_runtime.h>
#include <math.h>

// ---------------- constants ----------------
constexpr int   HD    = 512;          // hidden/emb dim
constexpr int   BATCH = 16;
constexpr int   SEQ   = 128;
constexpr int   VOC   = 10000;
constexpr int   VP    = 10112;        // vocab padded to multiple of 128
constexpr int   NROW  = BATCH * SEQ;  // 2048
constexpr int   MATSZ = HD * HD;
#define EPSF  1e-7f
#define MAXNF (1.0f - 1e-5f)

typedef __bf16 bf16;
typedef __attribute__((ext_vector_type(16))) __bf16 bf16x16;
typedef __attribute__((ext_vector_type(8)))  __bf16 bf16x8;
typedef __attribute__((ext_vector_type(8)))  float  f32x8;

// ---------------- small device helpers ----------------
__device__ __forceinline__ float sigmoidf_(float x) { return 1.0f / (1.0f + expf(-x)); }

// A-operand fragment for V_WMMA_F32_16X16X32_BF16.
// ISA layout: lane L: M=L%16, K = kh..kh+7 and 16+kh..16+kh+7 with kh = 8*(L>>4).
__device__ __forceinline__ bf16x16 load_afrag(const bf16* A, int m, int k0, int kh) {
    bf16x8 a0 = *(const bf16x8*)(A + m * HD + k0 + kh);
    bf16x8 a1 = *(const bf16x8*)(A + m * HD + k0 + 16 + kh);
    bf16x16 av;
#pragma unroll
    for (int j = 0; j < 8; ++j) { av[j] = a0[j]; av[j + 8] = a1[j]; }
    return av;
}

__device__ __forceinline__ bf16x16 load_bfrag(const bf16* B, size_t ld, int k0, int n, int kh) {
    bf16x16 bv;
#pragma unroll
    for (int j = 0; j < 8; ++j) {
        bv[j] = B[(size_t)(k0 + kh + j) * ld + n];
        bv[j + 8] = B[(size_t)(k0 + 16 + kh + j) * ld + n];
    }
    return bv;
}

// mobius_add on an owned 8-element slice given the row-scalars uv,uu,vv
__device__ __forceinline__ void mobius8(const float* u, const float* v,
                                        float uv, float uu, float vv, float* out) {
    float cu  = 1.0f + 2.0f * uv + vv;
    float cv  = 1.0f - uu;
    float den = 1.0f + 2.0f * uv + uu * vv;
    float inv = 1.0f / den;
#pragma unroll
    for (int j = 0; j < 8; ++j) out[j] = (cu * u[j] + cv * v[j]) * inv;
}

// Block-wide per-row reduction (16 rows x 64 threads/row) of up to 4 values.
// wave32 shfl tree -> 2 wave partials per row -> LDS combine. 2 barriers total.
__device__ __forceinline__ void rowreduce(float* sW /*32*4*/, float* sR /*16*4*/,
                                          int m, int sub, const float* in, float* out, int cnt) {
    float v[4];
#pragma unroll
    for (int c = 0; c < 4; ++c) v[c] = (c < cnt) ? in[c] : 0.0f;
#pragma unroll
    for (int mk = 1; mk < 32; mk <<= 1) {
#pragma unroll
        for (int c = 0; c < 4; ++c) if (c < cnt) v[c] += __shfl_xor(v[c], mk, 32);
    }
    const int w = threadIdx.x >> 5;
    if ((threadIdx.x & 31) == 0) {
#pragma unroll
        for (int c = 0; c < 4; ++c) if (c < cnt) sW[w * 4 + c] = v[c];
    }
    __syncthreads();
    if (sub < cnt) sR[m * 4 + sub] = sW[(2 * m) * 4 + sub] + sW[(2 * m + 1) * 4 + sub];
    __syncthreads();
#pragma unroll
    for (int c = 0; c < 4; ++c) if (c < cnt) out[c] = sR[m * 4 + c];
}

// ---------------- GEMM: [16,512](LDS bf16) x [512,512](global bf16, B layout) -> [16,512](LDS f32)
// One wave per 16-column tile; 32 waves cover N=512; 16 WMMA per wave.
// Software-pipelined: frag kb+1 is loaded before the WMMA on frag kb, so the
// scoreboard emits partial waits and loads overlap matrix math.
__device__ __forceinline__ void gemm16(const bf16* __restrict__ A,
                                       const bf16* __restrict__ B,
                                       float* __restrict__ O) {
    const int tid  = threadIdx.x;
    const int wv   = tid >> 5;
    const int lane = tid & 31;
    const int m    = lane & 15;
    const int kh   = (lane >> 4) << 3;
    const int nc0  = wv << 4;
    f32x8 acc = {0.f, 0.f, 0.f, 0.f, 0.f, 0.f, 0.f, 0.f};
    bf16x16 av = load_afrag(A, m, 0, kh);
    bf16x16 bv = load_bfrag(B, HD, 0, nc0 + m, kh);
#pragma unroll 4
    for (int kb = 0; kb < 16; ++kb) {
        bf16x16 avc = av, bvc = bv;
        if (kb < 15) {
            const int k1 = (kb + 1) << 5;
            av = load_afrag(A, m, k1, kh);
            bv = load_bfrag(B, HD, k1, nc0 + m, kh);
            __builtin_prefetch(B + (size_t)(k1 + 96 + lane) * HD + nc0, 0, 1);
        }
        acc = __builtin_amdgcn_wmma_f32_16x16x32_bf16(false, avc, false, bvc,
                                                      (short)0, acc, false, false);
    }
    const int mb = (lane >> 4) << 3;
    const int nc = nc0 + (lane & 15);
#pragma unroll
    for (int i = 0; i < 8; ++i) O[(mb + i) * HD + nc] = acc[i];
}

// m2 = mobius(mobius(expmap0(O1_row), expmap0(O2_row)), bias)   (bias read from global)
__device__ __forceinline__ void mob2(const float* sO1, const float* sO2,
                                     const float* __restrict__ bias, float bb,
                                     float* sW, float* sR,
                                     int m, int sub, int col0, float* m2out) {
    float u[8], v[8];
#pragma unroll
    for (int j = 0; j < 8; ++j) { u[j] = sO1[m * HD + col0 + j]; v[j] = sO2[m * HD + col0 + j]; }
    float in2[2] = {0.f, 0.f};
#pragma unroll
    for (int j = 0; j < 8; ++j) { in2[0] += u[j] * u[j]; in2[1] += v[j] * v[j]; }
    float o2[2];
    rowreduce(sW, sR, m, sub, in2, o2, 2);
    float nu = sqrtf(o2[0] + EPSF), nv = sqrtf(o2[1] + EPSF);
    float tu = tanhf(nu), tv = tanhf(nv);
    float su = tu / nu, sv = tv / nv;
    float e1[8], e2[8];
    float duv = 0.f;
#pragma unroll
    for (int j = 0; j < 8; ++j) { e1[j] = u[j] * su; e2[j] = v[j] * sv; duv += e1[j] * e2[j]; }
    float uvO;
    rowreduce(sW, sR, m, sub, &duv, &uvO, 1);
    float m1[8];
    mobius8(e1, e2, uvO, tu * tu, tv * tv, m1);
    float bown[8];
#pragma unroll
    for (int j = 0; j < 8; ++j) bown[j] = bias[col0 + j];
    float in2b[2] = {0.f, 0.f};
#pragma unroll
    for (int j = 0; j < 8; ++j) { in2b[0] += m1[j] * bown[j]; in2b[1] += m1[j] * m1[j]; }
    float o2b[2];
    rowreduce(sW, sR, m, sub, in2b, o2b, 2);
    mobius8(m1, bown, o2b[0], o2b[1], bb, m2out);
}

// ---------------- prep kernels ----------------
__global__ void tconv_w_kernel(const float* __restrict__ src, bf16* __restrict__ dst) {
    size_t tot = (size_t)3 * MATSZ;
    for (size_t e = (size_t)blockIdx.x * blockDim.x + threadIdx.x; e < tot;
         e += (size_t)gridDim.x * blockDim.x) {
        size_t l = e / MATSZ, r = e - l * MATSZ;
        size_t k = r >> 9, n = r & 511;
        dst[l * 6 * MATSZ + (k << 9) + n] = (bf16)src[l * MATSZ + (n << 9) + k];
    }
}

__global__ void tconv_pa_kernel(const float* __restrict__ src, bf16* __restrict__ dst) {
    size_t tot = (size_t)HD * VP;
    for (size_t e = (size_t)blockIdx.x * blockDim.x + threadIdx.x; e < tot;
         e += (size_t)gridDim.x * blockDim.x) {
        size_t k = e / VP, v = e - k * VP;
        dst[e] = (v < (size_t)VOC) ? (bf16)src[v * HD + k] : (bf16)0.0f;
    }
}

__global__ void vstats_kernel(const float* __restrict__ P, const float* __restrict__ A,
                              float* __restrict__ pp, float* __restrict__ pa,
                              float* __restrict__ aa, float* __restrict__ lam) {
    int v = blockIdx.x, lane = threadIdx.x;
    float sp = 0.f, sa = 0.f, spa = 0.f;
    if (v < VOC) {
        for (int k = lane; k < HD; k += 32) {
            float p = P[(size_t)v * HD + k], a = A[(size_t)v * HD + k];
            sp += p * p; sa += a * a; spa += p * a;
        }
    }
#pragma unroll
    for (int mk = 1; mk < 32; mk <<= 1) {
        sp += __shfl_xor(sp, mk, 32);
        sa += __shfl_xor(sa, mk, 32);
        spa += __shfl_xor(spa, mk, 32);
    }
    if (lane == 0) { pp[v] = sp; pa[v] = spa; aa[v] = sqrtf(sa + EPSF); lam[v] = 2.0f / (1.0f - sp); }
}

__global__ void embed_kernel(const float* __restrict__ E, const int* __restrict__ inp,
                             float* __restrict__ xo) {
    int blk = blockIdx.x;               // s*BATCH + b
    int s = blk >> 4, b = blk & 15;
    int idx = inp[b * SEQ + s];
    const float* src = E + (size_t)idx * HD;
    float* dst = xo + ((size_t)s * BATCH + b) * HD;
    for (int t = threadIdx.x; t < HD; t += blockDim.x) dst[t] = src[t];
}

// ---------------- hyperbolic GRU: one persistent block per layer ----------------
__global__ __launch_bounds__(1024) void gru_kernel(const float* __restrict__ xin,
                                                   float* __restrict__ xout,
                                                   const bf16* __restrict__ WtL,
                                                   const float* __restrict__ br,
                                                   const float* __restrict__ bz,
                                                   const float* __restrict__ bh) {
    extern __shared__ char smem[];
    bf16*  sA0 = (bf16*)smem;                    // 16x512 bf16 (lh / r*lh)
    bf16*  sA1 = sA0 + 16 * HD;                  // 16x512 bf16 (lx)
    float* sO1 = (float*)(sA1 + 16 * HD);        // 16x512 f32
    float* sO2 = sO1 + 16 * HD;                  // 16x512 f32
    float* sW  = sO2 + 16 * HD;                  // 32*4 wave partials
    float* sR  = sW + 32 * 4;                    // 16*4 row scalars
    float* sScl = sR + 16 * 4;                   // bias norms

    const int tid  = threadIdx.x;
    const int m    = tid >> 6;          // owned batch row 0..15
    const int sub  = tid & 63;
    const int col0 = sub << 3;          // owned 8 columns

    // bias squared-norms (once per layer)
    {
        float p0 = 0.f, p1 = 0.f, p2 = 0.f;
        if (tid < HD) { float x = br[tid], y = bz[tid], z = bh[tid]; p0 = x * x; p1 = y * y; p2 = z * z; }
        float v[4] = {p0, p1, p2, 0.f};
#pragma unroll
        for (int mk = 1; mk < 32; mk <<= 1) {
            v[0] += __shfl_xor(v[0], mk, 32);
            v[1] += __shfl_xor(v[1], mk, 32);
            v[2] += __shfl_xor(v[2], mk, 32);
        }
        if ((tid & 31) == 0) { int w = tid >> 5; sW[w * 4] = v[0]; sW[w * 4 + 1] = v[1]; sW[w * 4 + 2] = v[2]; }
        __syncthreads();
        if (tid == 0) {
            float s0 = 0.f, s1 = 0.f, s2 = 0.f;
            for (int w = 0; w < 32; ++w) { s0 += sW[w * 4]; s1 += sW[w * 4 + 1]; s2 += sW[w * 4 + 2]; }
            sScl[0] = s0; sScl[1] = s1; sScl[2] = s2;
        }
        __syncthreads();
    }
    const float bbr = sScl[0], bbz = sScl[1], bbh = sScl[2];

    float hx[8];
#pragma unroll
    for (int j = 0; j < 8; ++j) hx[j] = 0.0f;

#pragma unroll 1
    for (int t = 0; t < SEQ; ++t) {
        __syncthreads();
        // ---- lh = logmap0(h), lx = logmap0(xt); stage bf16 A operands
        float xv[8];
#pragma unroll
        for (int j = 0; j < 8; ++j) xv[j] = xin[((size_t)t * BATCH + m) * HD + col0 + j];
        float in2[2] = {0.f, 0.f};
#pragma unroll
        for (int j = 0; j < 8; ++j) { in2[0] += hx[j] * hx[j]; in2[1] += xv[j] * xv[j]; }
        float o2[2];
        rowreduce(sW, sR, m, sub, in2, o2, 2);
        const float hn2 = o2[0];
        float nh = sqrtf(o2[0] + EPSF), nx = sqrtf(o2[1] + EPSF);
        const float sch = atanhf(fminf(nh, MAXNF)) / nh;   // lh[j] == hx[j]*sch
        float scx = atanhf(fminf(nx, MAXNF)) / nx;
#pragma unroll
        for (int j = 0; j < 8; ++j) {
            sA0[m * HD + col0 + j] = (bf16)(hx[j] * sch);
            sA1[m * HD + col0 + j] = (bf16)(xv[j] * scx);
        }
        __syncthreads();

        // ---- r gate
        gemm16(sA0, WtL + 0 * MATSZ, sO1);   // lh @ Wr.T
        gemm16(sA1, WtL + 3 * MATSZ, sO2);   // lx @ Ur.T
        __syncthreads();
        float m2[8];
        mob2(sO1, sO2, br, bbr, sW, sR, m, sub, col0, m2);
        {
            float nn = 0.f;
#pragma unroll
            for (int j = 0; j < 8; ++j) nn += m2[j] * m2[j];
            float g2; rowreduce(sW, sR, m, sub, &nn, &g2, 1);
            float ng = sqrtf(g2 + EPSF);
            float ls = atanhf(fminf(ng, MAXNF)) / ng;
            // stage r*lh (r only needed here); overwrite sA0 after all reads done
            __syncthreads();
#pragma unroll
            for (int j = 0; j < 8; ++j)
                sA0[m * HD + col0 + j] = (bf16)(sigmoidf_(ls * m2[j]) * hx[j] * sch);
        }

        // ---- htil GEMMs (use r*lh while it is staged)
        __syncthreads();
        gemm16(sA0, WtL + 2 * MATSZ, sO1);   // (r*lh) @ Wh.T
        gemm16(sA1, WtL + 5 * MATSZ, sO2);   // lx @ Uh.T
        __syncthreads();
        float htil[8];
        mob2(sO1, sO2, bh, bbh, sW, sR, m, sub, col0, htil);

        // ---- z gate: rebuild lh in sA0 (cheap: hx*sch), then Wz/Uz GEMMs
        __syncthreads();
#pragma unroll
        for (int j = 0; j < 8; ++j) sA0[m * HD + col0 + j] = (bf16)(hx[j] * sch);
        __syncthreads();
        gemm16(sA0, WtL + 1 * MATSZ, sO1);   // lh @ Wz.T
        gemm16(sA1, WtL + 4 * MATSZ, sO2);   // lx @ Uz.T
        __syncthreads();
        float zg[8];
        mob2(sO1, sO2, bz, bbz, sW, sR, m, sub, col0, m2);
        {
            float nn = 0.f;
#pragma unroll
            for (int j = 0; j < 8; ++j) nn += m2[j] * m2[j];
            float g2; rowreduce(sW, sR, m, sub, &nn, &g2, 1);
            float ng = sqrtf(g2 + EPSF);
            float ls = atanhf(fminf(ng, MAXNF)) / ng;
#pragma unroll
            for (int j = 0; j < 8; ++j) zg[j] = sigmoidf_(ls * m2[j]);
        }

        // ---- delta = mobius(-h, htil); h = proj(mobius(h, expmap0(z*logmap0(delta))))
        float din[2] = {0.f, 0.f};
#pragma unroll
        for (int j = 0; j < 8; ++j) { din[0] += -hx[j] * htil[j]; din[1] += htil[j] * htil[j]; }
        float dout2[2];
        rowreduce(sW, sR, m, sub, din, dout2, 2);
        float del[8];
        {
            float mu[8];
#pragma unroll
            for (int j = 0; j < 8; ++j) mu[j] = -hx[j];
            mobius8(mu, htil, dout2[0], hn2, dout2[1], del);
        }
        float sd = 0.f;
#pragma unroll
        for (int j = 0; j < 8; ++j) sd += del[j] * del[j];
        float nd2; rowreduce(sW, sR, m, sub, &sd, &nd2, 1);
        float nd = sqrtf(nd2 + EPSF);
        float lsd = atanhf(fminf(nd, MAXNF)) / nd;
        float zld[8]; float sz = 0.f;
#pragma unroll
        for (int j = 0; j < 8; ++j) { zld[j] = zg[j] * del[j] * lsd; sz += zld[j] * zld[j]; }
        float nz2; rowreduce(sW, sR, m, sub, &sz, &nz2, 1);
        float nz = sqrtf(nz2 + EPSF);
        float tz = tanhf(nz);
        float sce = tz / nz;
        float ev[8]; float uvh = 0.f;
#pragma unroll
        for (int j = 0; j < 8; ++j) { ev[j] = zld[j] * sce; uvh += hx[j] * ev[j]; }
        float uvhO; rowreduce(sW, sR, m, sub, &uvh, &uvhO, 1);
        float hn[8];
        mobius8(hx, ev, uvhO, hn2, tz * tz, hn);
        float sp = 0.f;
#pragma unroll
        for (int j = 0; j < 8; ++j) sp += hn[j] * hn[j];
        float np2; rowreduce(sW, sR, m, sub, &sp, &np2, 1);
        float np = sqrtf(np2 + EPSF);
        float psc = fminf(1.0f, MAXNF / np);
#pragma unroll
        for (int j = 0; j < 8; ++j) {
            hx[j] = hn[j] * psc;
            xout[((size_t)t * BATCH + m) * HD + col0 + j] = hx[j];
        }
    }
}

// ---------------- X -> bf16 + row squared norms ----------------
__global__ void x2bf_kernel(const float* __restrict__ xin, bf16* __restrict__ X16,
                            float* __restrict__ xx) {
    int n = blockIdx.x, lane = threadIdx.x;
    int b = n >> 7, s = n & 127;
    const float* src = xin + ((size_t)s * BATCH + b) * HD;
    float ss = 0.f;
    for (int k = lane; k < HD; k += 32) {
        float x = src[k];
        X16[(size_t)n * HD + k] = (bf16)x;
        ss += x * x;
    }
#pragma unroll
    for (int mk = 1; mk < 32; mk <<= 1) ss += __shfl_xor(ss, mk, 32);
    if (lane == 0) xx[n] = ss;
}

// ---------------- fused hyperbolic-MLR logits + online softmax + NLL ----------------
__global__ __launch_bounds__(256) void logits_nll_kernel(
    const bf16* __restrict__ X16, const float* __restrict__ xx,
    const bf16* __restrict__ Pt, const bf16* __restrict__ At,
    const float* __restrict__ pp, const float* __restrict__ pa,
    const float* __restrict__ aa, const float* __restrict__ lam,
    const int* __restrict__ tgt, float* __restrict__ nll) {
    __shared__ bf16  sX[16 * HD];
    __shared__ float sxx[16];
    __shared__ int   stg[16];
    __shared__ float sM[8 * 16 * 3];

    const int tid = threadIdx.x;
    const int r0  = blockIdx.x * 16;

    // stage X tile via CDNA5 async DMA loads to LDS (ASYNCcnt-tracked)
    {
        const int i0 = tid * 32;                               // 32 bf16 = 64B per thread
        unsigned lbase = (unsigned)(uintptr_t)(sX + i0);       // low 32b of shared aperture = LDS offset
        unsigned long long gbase = (unsigned long long)(uintptr_t)(X16 + (size_t)r0 * HD + i0);
#pragma unroll
        for (int c = 0; c < 4; ++c) {
            unsigned laddr = lbase + c * 16;
            unsigned long long ga = gbase + c * 16;
            asm volatile("global_load_async_to_lds_b128 %0, %1, off"
                         :: "v"(laddr), "v"(ga) : "memory");
        }
        asm volatile("s_wait_asynccnt 0x0" ::: "memory");
    }
    if (tid < 16) { sxx[tid] = xx[r0 + tid]; stg[tid] = tgt[r0 + tid]; }
    __syncthreads();

    const int wv = tid >> 5, lane = tid & 31;
    const int m = lane & 15, kh = (lane >> 4) << 3, mb = (lane >> 4) << 3;
    const float NEGI = -__builtin_inff();
    float pm[8], ps[8], ptl[8];
#pragma unroll
    for (int i = 0; i < 8; ++i) { pm[i] = NEGI; ps[i] = 0.f; ptl[i] = NEGI; }

    for (int v0 = 0; v0 < VP; v0 += 128) {
        const int vt = v0 + (wv << 4);
        f32x8 aP = {0.f, 0.f, 0.f, 0.f, 0.f, 0.f, 0.f, 0.f};
        f32x8 aA = {0.f, 0.f, 0.f, 0.f, 0.f, 0.f, 0.f, 0.f};
        // software-pipelined dual-stream K loop
        bf16x16 av = load_afrag(sX, m, 0, kh);
        bf16x16 bp = load_bfrag(Pt, VP, 0, vt + m, kh);
        bf16x16 ba = load_bfrag(At, VP, 0, vt + m, kh);
#pragma unroll 4
        for (int kb = 0; kb < 16; ++kb) {
            bf16x16 avc = av, bpc = bp, bac = ba;
            if (kb < 15) {
                const int k1 = (kb + 1) << 5;
                av = load_afrag(sX, m, k1, kh);
                bp = load_bfrag(Pt, VP, k1, vt + m, kh);
                ba = load_bfrag(At, VP, k1, vt + m, kh);
            }
            aP = __builtin_amdgcn_wmma_f32_16x16x32_bf16(false, avc, false, bpc, (short)0, aP, false, false);
            aA = __builtin_amdgcn_wmma_f32_16x16x32_bf16(false, avc, false, bac, (short)0, aA, false, false);
        }
        const int col = vt + (lane & 15);
        const float ppv = pp[col], pav = pa[col], aav = aa[col], lmv = lam[col];
        const bool valid = col < VOC;
#pragma unroll
        for (int i = 0; i < 8; ++i) {
            const int row = mb + i;
            const float xxr = sxx[row];
            const float px = aP[i], xa = aA[i];
            const float alpha = 1.f - 2.f * px + xxr;
            const float beta  = 1.f - ppv;
            const float gamma = 1.f - 2.f * px + ppv * xxr;
            const float ig    = 1.f / gamma;
            const float da = (-alpha * pav + beta * xa) * ig;
            const float dd = (alpha * alpha * ppv - 2.f * alpha * beta * px + beta * beta * xxr) * ig * ig;
            const float arg = 2.f * da / ((1.f - dd) * aav + EPSF);
            const float lg  = lmv * aav * asinhf(arg);
            if (valid) {
                const float nm = fmaxf(pm[i], lg);
                ps[i] = ps[i] * expf(pm[i] - nm) + expf(lg - nm);
                pm[i] = nm;
                if (col == stg[row]) ptl[i] = fmaxf(ptl[i], lg);
            }
        }
    }
    // merge across the 16 lanes holding a row group (stay within half-wave)
#pragma unroll
    for (int i = 0; i < 8; ++i) {
        for (int mk = 1; mk < 16; mk <<= 1) {
            float om = __shfl_xor(pm[i], mk, 32);
            float os = __shfl_xor(ps[i], mk, 32);
            float ot = __shfl_xor(ptl[i], mk, 32);
            float nm = fmaxf(pm[i], om);
            ps[i] = ps[i] * expf(pm[i] - nm) + os * expf(om - nm);
            pm[i] = nm;
            ptl[i] = fmaxf(ptl[i], ot);
        }
    }
    if ((lane & 15) == 0) {
#pragma unroll
        for (int i = 0; i < 8; ++i) {
            int row = mb + i;
            sM[(wv * 16 + row) * 3 + 0] = pm[i];
            sM[(wv * 16 + row) * 3 + 1] = ps[i];
            sM[(wv * 16 + row) * 3 + 2] = ptl[i];
        }
    }
    __syncthreads();
    if (tid < 16) {
        float M = NEGI, S = 0.f, T = NEGI;
        for (int w = 0; w < 8; ++w) {
            float om = sM[(w * 16 + tid) * 3 + 0];
            float os = sM[(w * 16 + tid) * 3 + 1];
            float ot = sM[(w * 16 + tid) * 3 + 2];
            float nm = fmaxf(M, om);
            S = S * expf(M - nm) + os * expf(om - nm);
            M = nm;
            T = fmaxf(T, ot);
        }
        nll[r0 + tid] = (M + logf(S)) - T;
    }
}

__global__ void mean_kernel(const float* __restrict__ nll, float* __restrict__ out) {
    __shared__ float sm[256];
    float s = 0.f;
    for (int i = threadIdx.x; i < NROW; i += 256) s += nll[i];
    sm[threadIdx.x] = s;
    __syncthreads();
    for (int st = 128; st > 0; st >>= 1) {
        if (threadIdx.x < st) sm[threadIdx.x] += sm[threadIdx.x + st];
        __syncthreads();
    }
    if (threadIdx.x == 0) out[0] = sm[0] / (float)NROW;
}

// ---------------- launcher ----------------
extern "C" void kernel_launch(void* const* d_in, const int* in_sizes, int n_in,
                              void* d_out, int out_size, void* d_ws, size_t ws_size,
                              hipStream_t stream) {
    (void)in_sizes; (void)n_in; (void)out_size; (void)ws_size;
    const int*   inp = (const int*)d_in[0];
    const int*   tgt = (const int*)d_in[1];
    const float* E   = (const float*)d_in[2];
    const float* Wr  = (const float*)d_in[3];
    const float* Wz  = (const float*)d_in[4];
    const float* Wh  = (const float*)d_in[5];
    const float* Ur  = (const float*)d_in[6];
    const float* Uz  = (const float*)d_in[7];
    const float* Uh  = (const float*)d_in[8];
    const float* br  = (const float*)d_in[9];
    const float* bz  = (const float*)d_in[10];
    const float* bh  = (const float*)d_in[11];
    const float* P   = (const float*)d_in[12];
    const float* A   = (const float*)d_in[13];

    char* ws = (char*)d_ws;
    size_t off = 0;
    auto alloc = [&](size_t b) { size_t o = off; off += (b + 255) & ~(size_t)255; return o; };
    bf16*  Wt  = (bf16*)(ws + alloc(sizeof(bf16) * 18 * MATSZ));
    bf16*  Pt  = (bf16*)(ws + alloc(sizeof(bf16) * HD * VP));
    bf16*  At  = (bf16*)(ws + alloc(sizeof(bf16) * HD * VP));
    float* pp  = (float*)(ws + alloc(sizeof(float) * VP));
    float* pa  = (float*)(ws + alloc(sizeof(float) * VP));
    float* aa  = (float*)(ws + alloc(sizeof(float) * VP));
    float* lam = (float*)(ws + alloc(sizeof(float) * VP));
    float* xA  = (float*)(ws + alloc(sizeof(float) * SEQ * BATCH * HD));
    float* xB  = (float*)(ws + alloc(sizeof(float) * SEQ * BATCH * HD));
    bf16*  X16 = (bf16*)(ws + alloc(sizeof(bf16) * NROW * HD));
    float* xx  = (float*)(ws + alloc(sizeof(float) * NROW));
    float* nll = (float*)(ws + alloc(sizeof(float) * NROW));

    tconv_w_kernel<<<1024, 256, 0, stream>>>(Wr, Wt + 0 * MATSZ);
    tconv_w_kernel<<<1024, 256, 0, stream>>>(Wz, Wt + 1 * MATSZ);
    tconv_w_kernel<<<1024, 256, 0, stream>>>(Wh, Wt + 2 * MATSZ);
    tconv_w_kernel<<<1024, 256, 0, stream>>>(Ur, Wt + 3 * MATSZ);
    tconv_w_kernel<<<1024, 256, 0, stream>>>(Uz, Wt + 4 * MATSZ);
    tconv_w_kernel<<<1024, 256, 0, stream>>>(Uh, Wt + 5 * MATSZ);
    tconv_pa_kernel<<<2048, 256, 0, stream>>>(P, Pt);
    tconv_pa_kernel<<<2048, 256, 0, stream>>>(A, At);
    vstats_kernel<<<VP, 32, 0, stream>>>(P, A, pp, pa, aa, lam);

    embed_kernel<<<SEQ * BATCH, 256, 0, stream>>>(E, inp, xA);

    constexpr size_t GRU_SMEM = 2 * 16 * HD * sizeof(bf16) + 2 * 16 * HD * sizeof(float) +
                                (32 * 4 + 16 * 4 + 8) * sizeof(float);
    gru_kernel<<<1, 1024, GRU_SMEM, stream>>>(xA, xB, Wt + (size_t)0 * 6 * MATSZ,
                                              br + 0 * HD, bz + 0 * HD, bh + 0 * HD);
    gru_kernel<<<1, 1024, GRU_SMEM, stream>>>(xB, xA, Wt + (size_t)1 * 6 * MATSZ,
                                              br + 1 * HD, bz + 1 * HD, bh + 1 * HD);
    gru_kernel<<<1, 1024, GRU_SMEM, stream>>>(xA, xB, Wt + (size_t)2 * 6 * MATSZ,
                                              br + 2 * HD, bz + 2 * HD, bh + 2 * HD);

    x2bf_kernel<<<NROW, 32, 0, stream>>>(xB, X16, xx);
    logits_nll_kernel<<<NROW / 16, 256, 0, stream>>>(X16, xx, Pt, At, pp, pa, aa, lam, tgt, nll);
    mean_kernel<<<1, 256, 0, stream>>>(nll, (float*)d_out);
}
